// LF_BF_QP_Net_86758339379732
// MI455X (gfx1250) — compile-verified, matrix operand order checked by reading, not run on hardware
//
#include <hip/hip_runtime.h>

// ---------------------------------------------------------------------------
// CDNA5 (gfx1250) fused CLF/CBF network kernel.
// Four (B x 256) @ (256 x 256) GEMMs on v_wmma_f32_16x16x32_f16, software-
// pipelined LDS fragment loads, hardware v_tanh for activations.
// ---------------------------------------------------------------------------

typedef __attribute__((ext_vector_type(16))) _Float16 v16h;
typedef __attribute__((ext_vector_type(8)))  _Float16 v8h;
typedef __attribute__((ext_vector_type(8)))  float    v8f;

union F16Frag { v16h v; v8h h[2]; };

#define WST 264   // LDS weight row stride in halfs (16B-aligned rows, padded)
#define AST 264   // LDS activation row stride in halfs
#define ROWS_PER_BLOCK 128
#define THREADS 256

// ---- fast tanh: hardware v_tanh_f32 on gfx1250 if available ----
__device__ __forceinline__ float fast_tanh(float x) {
#if __has_builtin(__builtin_amdgcn_tanhf)
  return __builtin_amdgcn_tanhf(x);
#elif __has_builtin(__builtin_amdgcn_tanh_f32)
  return __builtin_amdgcn_tanh_f32(x);
#else
  // tanh(x) = sign(x) * (1 - e) / (1 + e),  e = exp(-2|x|) via v_exp_f32
  float ax = __builtin_fabsf(x);
  float e  = __builtin_amdgcn_exp2f(ax * -2.885390082f);   // 2^(-2|x|*log2 e)
  float t  = (1.0f - e) * __builtin_amdgcn_rcpf(1.0f + e);
  return __builtin_copysignf(t, x);
#endif
}

// One wave computes D(16x256) = A(16x256) * B(256x256) with 128 WMMAs.
// Flattened (ks,nt) loop, rotating 3-buffer B prefetch (depth 2), double-
// buffered A, and a rebased high pointer so all ds offsets fit 16-bit imm.
__device__ __forceinline__ void wave_gemm_16x256x256(
    const _Float16* __restrict__ aLane, const _Float16* __restrict__ bLane,
    int grp, v8f* acc)
{
  const _Float16* bHi = bLane + (8 * 16 * WST);  // tiles 8..15 rebased
  const int kg = grp * 8;

  auto bAddr = [&](int i) -> const _Float16* {
    const int ks = i >> 4, nt = i & 15;
    const _Float16* base = (nt < 8) ? bLane : bHi;
    return base + (nt & 7) * (16 * WST) + ks * 32 + kg;
  };

  F16Frag a0, a1;
  F16Frag bf[3];

  a0.h[0] = *(const v8h*)(aLane + kg);
  a0.h[1] = *(const v8h*)(aLane + kg + 16);
  {
    const _Float16* p = bAddr(0);
    bf[0].h[0] = *(const v8h*)p;  bf[0].h[1] = *(const v8h*)(p + 16);
  }
  {
    const _Float16* p = bAddr(1);
    bf[1].h[0] = *(const v8h*)p;  bf[1].h[1] = *(const v8h*)(p + 16);
  }

#pragma unroll
  for (int i = 0; i < 128; ++i) {
    const int ks = i >> 4, nt = i & 15;
    if (i + 2 < 128) {                       // prefetch B two steps ahead
      const _Float16* p = bAddr(i + 2);
      F16Frag& nb = bf[(i + 2) % 3];
      nb.h[0] = *(const v8h*)p;
      nb.h[1] = *(const v8h*)(p + 16);
    }
    if (nt == 12 && ks < 7) {                // prefetch next k-step's A
      const int k0n = (ks + 1) * 32 + kg;
      F16Frag& na = (ks & 1) ? a0 : a1;
      na.h[0] = *(const v8h*)(aLane + k0n);
      na.h[1] = *(const v8h*)(aLane + k0n + 16);
    }
    const F16Frag& ca = (ks & 1) ? a1 : a0;
    acc[nt] = __builtin_amdgcn_wmma_f32_16x16x32_f16(
        false, ca.v, false, bf[i % 3].v, (short)0, acc[nt], false, false);
  }
}

__global__ __launch_bounds__(THREADS)
void lf_bf_qp_net_kernel(
    const float* __restrict__ x,
    const float* __restrict__ VW1, const float* __restrict__ Vb1,
    const float* __restrict__ VW2, const float* __restrict__ Vb2,
    const float* __restrict__ HW1, const float* __restrict__ Hb1,
    const float* __restrict__ HW2, const float* __restrict__ Hb2,
    const float* __restrict__ HW3, const float* __restrict__ Hb3,
    const float* __restrict__ uW1, const float* __restrict__ ub1,
    const float* __restrict__ uW2, const float* __restrict__ ub2,
    float* __restrict__ out, int B)
{
  __shared__ _Float16 wbuf[256 * WST];            // 132 KB: big weight (fwd / bwd-T)
  __shared__ _Float16 abuf[ROWS_PER_BLOCK * AST]; //  66 KB: layer-1 acts, then s_h
  __shared__ float xs[ROWS_PER_BLOCK * 2];
  __shared__ float w1s[256 * 2];
  __shared__ float b1s[256];
  __shared__ float b2s[256];
  __shared__ float w3s[256];
  __shared__ float rb[8][ROWS_PER_BLOCK]; // 0:u 1:V 2:gv0 3:gv1 4:H 5:(1-H^2) 6:gh0 7:gh1

  const int tid  = threadIdx.x;
  const int wave = tid >> 5;
  const int lane = tid & 31;
  const int sub  = lane & 15;
  const int grp  = lane >> 4;
  const int rowBaseL = wave * 16;
  const int blockRow = blockIdx.x * ROWS_PER_BLOCK;

  const v8f vzero = {0.f, 0.f, 0.f, 0.f, 0.f, 0.f, 0.f, 0.f};

  // ---- x tile ----
  for (int i = tid; i < ROWS_PER_BLOCK * 2; i += THREADS)
    xs[i] = x[blockRow * 2 + i];

  // =========================================================================
  // Two identical network passes: V-net (phase 0) and H-net (phase 1).
  // =========================================================================
#pragma unroll 1
  for (int net = 0; net < 2; ++net) {
    const float* W1 = net ? HW1 : VW1;
    const float* B1 = net ? Hb1 : Vb1;
    const float* W2 = net ? HW2 : VW2;
    const float* B2 = net ? Hb2 : Vb2;

    // small weights + biases + head vector
    for (int i = tid; i < 512; i += THREADS) w1s[i] = W1[i];
    for (int i = tid; i < 256; i += THREADS) {
      b1s[i] = B1[i];
      b2s[i] = B2[i];
      w3s[i] = net ? HW3[i] : 0.0f;
    }
    // big weight, row-major f16: wbuf[h_out][k_in]
    for (int i = tid; i < 65536; i += THREADS)
      wbuf[(i >> 8) * WST + (i & 255)] = (_Float16)W2[i];
    __syncthreads();

    // layer-1 elementwise: a1 = tanh(x @ W1^T + b1)  (K = 2, scalar)
    for (int i = tid; i < ROWS_PER_BLOCK * 256; i += THREADS) {
      int r = i >> 8, h = i & 255;
      float p = fmaf(xs[r * 2], w1s[h * 2],
                fmaf(xs[r * 2 + 1], w1s[h * 2 + 1], b1s[h]));
      abuf[r * AST + h] = (_Float16)fast_tanh(p);
    }
    __syncthreads();

    // ---- forward GEMM: pre2 = a1 @ W2^T ----
    {
      v8f acc[16];
#pragma unroll
      for (int nt = 0; nt < 16; ++nt) acc[nt] = vzero;
      wave_gemm_16x256x256(abuf + (rowBaseL + sub) * AST, wbuf + sub * WST, grp, acc);

      float part[8] = {0.f, 0.f, 0.f, 0.f, 0.f, 0.f, 0.f, 0.f};
#pragma unroll
      for (int nt = 0; nt < 16; ++nt) {
        const int col = nt * 16 + sub;
        const float bb  = b2s[col];
        const float w3v = w3s[col];
#pragma unroll
        for (int r = 0; r < 8; ++r) {
          const int rl = rowBaseL + grp * 8 + r;
          float a2 = fast_tanh(acc[nt][r] + bb);
          float d  = 1.0f - a2 * a2;
          float s, p;
          if (net == 0) { s = a2 * d;  p = a2 * a2; }   // V: s_h=V2(1-V2^2), V=0.5*sum V2^2
          else          { s = w3v * d; p = a2 * w3v; }  // H: s_h=W3(1-H2^2), head pre-act
          abuf[rl * AST + col] = (_Float16)s;           // own rows only: safe
          part[r] += p;
        }
      }
#pragma unroll
      for (int r = 0; r < 8; ++r) {
        float s = part[r];
        s += __shfl_xor(s, 1); s += __shfl_xor(s, 2);
        s += __shfl_xor(s, 4); s += __shfl_xor(s, 8);
        if (sub == 0) {
          const int rl = rowBaseL + grp * 8 + r;
          if (net == 0) {
            rb[1][rl] = 0.5f * s;                       // V
          } else {
            float Hv = fast_tanh(s + Hb3[0]);           // H
            rb[4][rl] = Hv;
            rb[5][rl] = 1.0f - Hv * Hv;
          }
        }
      }
    }
    __syncthreads();

    // reload big weight TRANSPOSED: wbuf[k][h] = W2[h][k]
    for (int i = tid; i < 65536; i += THREADS)
      wbuf[(i & 255) * WST + (i >> 8)] = (_Float16)W2[i];
    __syncthreads();

    // ---- backward GEMM: bsum = s @ W2 ----
    {
      v8f acc[16];
#pragma unroll
      for (int nt = 0; nt < 16; ++nt) acc[nt] = vzero;
      wave_gemm_16x256x256(abuf + (rowBaseL + sub) * AST, wbuf + sub * WST, grp, acc);

      float g0[8] = {0.f, 0.f, 0.f, 0.f, 0.f, 0.f, 0.f, 0.f};
      float g1[8] = {0.f, 0.f, 0.f, 0.f, 0.f, 0.f, 0.f, 0.f};
#pragma unroll
      for (int nt = 0; nt < 16; ++nt) {
        const int k   = nt * 16 + sub;
        const float kw0 = w1s[k * 2];
        const float kw1 = w1s[k * 2 + 1];
        const float bb  = b1s[k];
#pragma unroll
        for (int r = 0; r < 8; ++r) {
          const int rl = rowBaseL + grp * 8 + r;
          // recompute layer-1 activation (abuf was overwritten with s_h)
          float a1 = fast_tanh(fmaf(xs[rl * 2], kw0, fmaf(xs[rl * 2 + 1], kw1, bb)));
          float t  = (1.0f - a1 * a1) * acc[nt][r];
          g0[r] = fmaf(t, kw0, g0[r]);
          g1[r] = fmaf(t, kw1, g1[r]);
        }
      }
#pragma unroll
      for (int r = 0; r < 8; ++r) {
        float s0 = g0[r], s1 = g1[r];
        s0 += __shfl_xor(s0, 1); s0 += __shfl_xor(s0, 2);
        s0 += __shfl_xor(s0, 4); s0 += __shfl_xor(s0, 8);
        s1 += __shfl_xor(s1, 1); s1 += __shfl_xor(s1, 2);
        s1 += __shfl_xor(s1, 4); s1 += __shfl_xor(s1, 8);
        if (sub == 0) {
          const int rl = rowBaseL + grp * 8 + r;
          rb[2 + 4 * net][rl] = s0;   // gv0 / gh0(pre)
          rb[3 + 4 * net][rl] = s1;   // gv1 / gh1(pre)
        }
      }
    }
    __syncthreads();
  }

  // ---- u network (pure per-row 256-dim matvec) ----
  for (int i = tid; i < 512; i += THREADS) w1s[i] = uW1[i];
  for (int i = tid; i < 256; i += THREADS) { b1s[i] = ub1[i]; w3s[i] = uW2[i]; }
  __syncthreads();
  {
    const int row  = tid >> 1;
    const int half = tid & 1;
    const float x0 = xs[row * 2], x1 = xs[row * 2 + 1];
    float a = 0.f;
    for (int h = half * 128; h < half * 128 + 128; ++h) {
      float u1 = fast_tanh(fmaf(x0, w1s[h * 2], fmaf(x1, w1s[h * 2 + 1], b1s[h])));
      a = fmaf(u1, w3s[h], a);
    }
    a += __shfl_xor(a, 1);
    if (half == 0) rb[0][row] = fast_tanh(a + ub2[0]);
  }
  __syncthreads();

  // ---- finalize: u, V, Vdot, H, Hdot ----
  if (tid < ROWS_PER_BLOCK) {
    const int row = tid;
    const int rg  = blockRow + row;
    const float x0 = xs[row * 2], x1 = xs[row * 2 + 1];
    const float f0 = x1;
    const float f1 = -sinf(x0) - 0.1f * x1;
    const float u  = rb[0][row];
    const float gv0 = rb[2][row], gv1 = rb[3][row];
    const float hs  = rb[5][row];
    const float gh0 = hs * rb[6][row], gh1 = hs * rb[7][row];
    out[rg]             = u;
    out[B + rg]         = rb[1][row];
    out[2 * B + rg]     = fmaf(gv1, u, fmaf(gv0, f0, gv1 * f1));  // Vdot
    out[3 * B + rg]     = rb[4][row];
    out[4 * B + rg]     = fmaf(gh1, u, fmaf(gh0, f0, gh1 * f1));  // Hdot
  }
}

extern "C" void kernel_launch(void* const* d_in, const int* in_sizes, int n_in,
                              void* d_out, int out_size, void* d_ws, size_t ws_size,
                              hipStream_t stream) {
  const float* x   = (const float*)d_in[0];
  const float* VW1 = (const float*)d_in[1];
  const float* Vb1 = (const float*)d_in[2];
  const float* VW2 = (const float*)d_in[3];
  const float* Vb2 = (const float*)d_in[4];
  const float* HW1 = (const float*)d_in[5];
  const float* Hb1 = (const float*)d_in[6];
  const float* HW2 = (const float*)d_in[7];
  const float* Hb2 = (const float*)d_in[8];
  const float* HW3 = (const float*)d_in[9];
  const float* Hb3 = (const float*)d_in[10];
  const float* uW1 = (const float*)d_in[11];
  const float* ub1 = (const float*)d_in[12];
  const float* uW2 = (const float*)d_in[13];
  const float* ub2 = (const float*)d_in[14];

  const int B = in_sizes[0] / 2;            // 131072
  const int blocks = B / ROWS_PER_BLOCK;    // 1024

  hipLaunchKernelGGL(lf_bf_qp_net_kernel, dim3(blocks), dim3(THREADS), 0, stream,
                     x, VW1, Vb1, VW2, Vb2, HW1, Hb1, HW2, Hb2, HW3, Hb3,
                     uW1, ub1, uW2, ub2, (float*)d_out, B);
}
